// CosineAttention_45406394254083
// MI455X (gfx1250) — compile-verified, matrix operand order checked by reading, not run on hardware
//
#include <hip/hip_runtime.h>
#include <hip/hip_bf16.h>
#include <stdint.h>

typedef _Float16 half_t;
typedef __attribute__((ext_vector_type(16))) _Float16 v16h;
typedef __attribute__((ext_vector_type(8)))  float    v8f;
typedef __attribute__((ext_vector_type(4)))  float    v4f;

#define B_   8
#define N_   1024
#define C_   768
#define H_   12
#define D_   64
#define QKV_ (3 * H_ * D_)   // 2304
#define NEG_MAX (-3.402823466e38f)

__device__ __forceinline__ v8f wmma_f16(v16h a, v16h b, v8f c) {
    return __builtin_amdgcn_wmma_f32_16x16x32_f16(
        false, a, false, b, (short)0, c, false, false);
}

__device__ __forceinline__ half_t w_lo(uint32_t w) {
    union { uint32_t u; half_t h[2]; } c; c.u = w; return c.h[0];
}
__device__ __forceinline__ half_t w_hi(uint32_t w) {
    union { uint32_t u; half_t h[2]; } c; c.u = w; return c.h[1];
}
__device__ __forceinline__ uint32_t pack2(half_t a, half_t b) {
    union { uint32_t u; half_t h[2]; } c; c.h[0] = a; c.h[1] = b; return c.u;
}

// ---------------------------------------------------------------------------
__global__ __launch_bounds__(256) void cvt_f32_f16(const float* __restrict__ src,
                                                   half_t* __restrict__ dst, int n) {
    int i = blockIdx.x * blockDim.x + threadIdx.x;
    if (i < n) dst[i] = (half_t)src[i];
}

// ---------------------------------------------------------------------------
// WMMA GEMM:  Cm[M,Nn] = A[M,K] @ Bm[Nn,K]^T (f16 in, OutT out)
// One wave -> 32(M) x 64(N) tile, K-step 32, ping-pong double-buffered regs.
// ---------------------------------------------------------------------------
template <int K, typename OutT>
__global__ __launch_bounds__(256) void gemm_wmma_ab(const half_t* __restrict__ A,
                                                    const half_t* __restrict__ Bm,
                                                    OutT* __restrict__ Cm,
                                                    int M, int Nn) {
    const int wave = blockIdx.x * (blockDim.x >> 5) + (threadIdx.x >> 5);
    const int lane = threadIdx.x & 31;
    const int nTiles = Nn >> 6;
    const int mTile = wave / nTiles;
    const int nTile = wave % nTiles;
    if (mTile >= (M >> 5)) return;

    const int m0 = mTile << 5;
    const int n0 = nTile << 6;
    const int lrow = lane & 15;
    const int lk   = (lane >> 4) << 4;

    const half_t* aP = A + (size_t)(m0 + lrow) * K + lk;
    const half_t* aQ = aP + (size_t)16 * K;
    const half_t* bP = Bm + (size_t)(n0 + lrow) * K + lk;
    const size_t  bS = (size_t)16 * K;

    v8f acc[8] = {};
    v16h a0[2], b0[4], a1[2], b1[4];
    auto ld = [&](int s, v16h* a, v16h* b) {
        const int kk = s * 32;
        a[0] = *(const v16h*)(aP + kk);
        a[1] = *(const v16h*)(aQ + kk);
        b[0] = *(const v16h*)(bP + kk);
        b[1] = *(const v16h*)(bP + bS + kk);
        b[2] = *(const v16h*)(bP + 2 * bS + kk);
        b[3] = *(const v16h*)(bP + 3 * bS + kk);
    };
    auto mm = [&](v16h* a, v16h* b) {
#pragma unroll
        for (int i = 0; i < 2; ++i)
#pragma unroll
            for (int j = 0; j < 4; ++j)
                acc[i * 4 + j] = wmma_f16(a[i], b[j], acc[i * 4 + j]);
    };

    constexpr int NK = K / 32;
    ld(0, a0, b0);
#pragma unroll
    for (int s = 0; s < NK; s += 2) {
        if (s + 1 < NK) ld(s + 1, a1, b1);
        mm(a0, b0);
        if (s + 2 < NK) ld(s + 2, a0, b0);
        if (s + 1 < NK) mm(a1, b1);
    }

    const int chi = (lane >> 4) << 3;
#pragma unroll
    for (int g = 0; g < 2; ++g) {
        OutT* cBase = Cm + (size_t)(m0 + g * 16 + chi) * Nn + n0 + lrow;
#pragma unroll
        for (int r = 0; r < 8; ++r) {
            OutT* cr = cBase + (size_t)r * Nn;
            cr[0]  = (OutT)acc[g * 4 + 0][r];
            cr[16] = (OutT)acc[g * 4 + 1][r];
            cr[32] = (OutT)acc[g * 4 + 2][r];
            cr[48] = (OutT)acc[g * 4 + 3][r];
        }
    }
}

// ---------------------------------------------------------------------------
// QKV prep: L2-normalize q/k, emit qh/kh row-major f16, vt transposed f16.
// ---------------------------------------------------------------------------
__global__ __launch_bounds__(256) void qkv_prep(const half_t* __restrict__ qkv,
                                                half_t* __restrict__ qh,
                                                half_t* __restrict__ kh,
                                                half_t* __restrict__ vt) {
    const int wave = blockIdx.x * (blockDim.x >> 5) + (threadIdx.x >> 5);
    const int lane = threadIdx.x & 31;
    if (wave >= B_ * H_ * N_) return;

    const int b   = wave / (H_ * N_);
    const int rem = wave % (H_ * N_);
    const int h   = rem / N_;
    const int n   = rem % N_;
    const int bh  = b * H_ + h;

    const uint32_t* p32 =
        (const uint32_t*)(qkv + (size_t)(b * N_ + n) * QKV_ + h * (D_ * 3));
    uint32_t w0 = p32[lane * 3 + 0];
    uint32_t w1 = p32[lane * 3 + 1];
    uint32_t w2 = p32[lane * 3 + 2];
    float q0 = (float)w_lo(w0), k0 = (float)w_hi(w0), v0 = (float)w_lo(w1);
    float q1 = (float)w_hi(w1), k1 = (float)w_lo(w2), v1 = (float)w_hi(w2);

    float sq = q0 * q0 + q1 * q1;
    float sk = k0 * k0 + k1 * k1;
#pragma unroll
    for (int m = 16; m >= 1; m >>= 1) {
        sq += __shfl_xor(sq, m, 32);
        sk += __shfl_xor(sk, m, 32);
    }
    const float rq = 1.0f / fmaxf(sqrtf(sq), 1e-12f);
    const float rk = 1.0f / fmaxf(sqrtf(sk), 1e-12f);

    const int d0 = lane * 2;
    const size_t ro = (size_t)(bh * N_ + n) * D_;
    *(uint32_t*)(qh + ro + d0) = pack2((half_t)(q0 * rq), (half_t)(q1 * rq));
    *(uint32_t*)(kh + ro + d0) = pack2((half_t)(k0 * rk), (half_t)(k1 * rk));

    const size_t vo = (size_t)bh * D_ * N_ + n;
    vt[vo + (size_t)d0 * N_]       = (half_t)v0;
    vt[vo + (size_t)(d0 + 1) * N_] = (half_t)v1;
}

// ---------------------------------------------------------------------------
// Flash attention, transposed-score (S^T = K Q^T), 32 output rows per wave.
// Lane owns rows iA = i0+lane%16 and iB = i0+16+lane%16.
// Single K buffer with in-place next-iteration prefetch; V loaded per body.
// Row groups processed sequentially to cap register pressure (< 256 VGPRs).
// ---------------------------------------------------------------------------
__global__ __launch_bounds__(256) void attn_flash(const half_t* __restrict__ qh,
                                                  const half_t* __restrict__ kh,
                                                  const half_t* __restrict__ vt,
                                                  const float* __restrict__ bias,
                                                  const uint8_t* __restrict__ mask,
                                                  const float* __restrict__ tptr,
                                                  half_t* __restrict__ attnh) {
    const int wid  = threadIdx.x >> 5;
    const int lane = threadIdx.x & 31;
    const int wave = blockIdx.x * 8 + wid;
    const int iTiles = N_ >> 5;                  // 32-row tiles
    if (wave >= B_ * H_ * iTiles) return;

    const int b   = wave / (H_ * iTiles);
    const int rem = wave % (H_ * iTiles);
    const int h   = rem / iTiles;
    const int i0  = (rem % iTiles) << 5;
    const int bh  = b * H_ + h;

    const int lcol = lane & 15;
    const int lhi  = lane >> 4;
    const float temp = tptr[0];

    const half_t*  kBase = kh + (size_t)bh * N_ * D_;
    const half_t*  vBase = vt + (size_t)bh * D_ * N_;
    const uint8_t* mPad  = mask + (size_t)b * N_;

    const int iA = i0 + lcol;
    const int iB = i0 + 16 + lcol;
    const float* bRowA = bias + ((size_t)h * N_ + iA) * N_;
    const float* bRowB = bias + ((size_t)h * N_ + iB) * N_;
    const bool rowOKA = (mPad[iA] == 0);
    const bool rowOKB = (mPad[iB] == 0);

    const half_t* qpA = qh + (size_t)bh * N_ * D_ + (size_t)iA * D_ + (lhi << 4);
    const half_t* qpB = qpA + (size_t)16 * D_;
    v16h qB0a = *(const v16h*)(qpA);
    v16h qB1a = *(const v16h*)(qpA + 32);
    v16h qB0b = *(const v16h*)(qpB);
    v16h qB1b = *(const v16h*)(qpB + 32);

    v8f oa0 = {}, oa1 = {}, oa2 = {}, oa3 = {};
    v8f ob0 = {}, ob1 = {}, ob2 = {}, ob3 = {};
    float mA = NEG_MAX, lA = 0.0f, mBv = NEG_MAX, lB = 0.0f;

    v16h kf[4];                                  // single K buffer
    auto loadK = [&](int j0) {
        const half_t* kp0 = kBase + (size_t)(j0 + lcol) * D_ + (lhi << 4);
        const half_t* kp1 = kp0 + (size_t)16 * D_;
        kf[0] = *(const v16h*)(kp0);
        kf[1] = *(const v16h*)(kp0 + 32);
        kf[2] = *(const v16h*)(kp1);
        kf[3] = *(const v16h*)(kp1 + 32);
    };

    // per-group score -> softmax -> PV step
    auto group = [&](v8f& ST0, v8f& ST1, const float* bRow, bool rowOK,
                     int iMine, int jb, uint32_t m0lo, uint32_t m0hi,
                     uint32_t m1lo, uint32_t m1hi, float& m_, float& l_,
                     v16h* vf, v8f& g0, v8f& g1, v8f& g2, v8f& g3) {
        v4f bz0a = *(const v4f*)(bRow + jb);
        v4f bz0b = *(const v4f*)(bRow + jb + 4);
        v4f bz1a = *(const v4f*)(bRow + jb + 16);
        v4f bz1b = *(const v4f*)(bRow + jb + 20);
        const int di0 = iMine - jb, di1 = di0 - 16;

        float p0[8], p1[8];
#pragma unroll
        for (int r = 0; r < 8; ++r) {
            const uint32_t w0 = (r < 4) ? m0lo : m0hi;
            const uint32_t w1 = (r < 4) ? m1lo : m1hi;
            const bool cOK0 = ((w0 >> (8 * (r & 3))) & 0xffu) == 0;
            const bool cOK1 = ((w1 >> (8 * (r & 3))) & 0xffu) == 0;
            float b0 = (r < 4) ? bz0a[r & 3] : bz0b[r & 3];
            float b1 = (r < 4) ? bz1a[r & 3] : bz1b[r & 3];
            p0[r] = (rowOK && cOK0 && r <= di0) ? fmaf(ST0[r], temp, b0) : NEG_MAX;
            p1[r] = (rowOK && cOK1 && r <= di1) ? fmaf(ST1[r], temp, b1) : NEG_MAX;
        }

        float mloc = NEG_MAX;
#pragma unroll
        for (int r = 0; r < 8; ++r) mloc = fmaxf(mloc, fmaxf(p0[r], p1[r]));
        mloc = fmaxf(mloc, __shfl_xor(mloc, 16, 32));
        const float mNew = fmaxf(m_, mloc);
        const float alpha = __expf(m_ - mNew);
        float sum = 0.0f;
#pragma unroll
        for (int r = 0; r < 8; ++r) {
            p0[r] = __expf(p0[r] - mNew);
            p1[r] = __expf(p1[r] - mNew);
            sum += p0[r] + p1[r];
        }
        sum += __shfl_xor(sum, 16, 32);
        l_ = l_ * alpha + sum;
        m_ = mNew;

#pragma unroll
        for (int r = 0; r < 8; ++r) {
            float aR = __shfl(alpha, r + (lhi << 3), 32);
            g0[r] *= aR; g1[r] *= aR; g2[r] *= aR; g3[r] *= aR;
        }

        union { v16h v; uint32_t u[8]; } pf;
#pragma unroll
        for (int r = 0; r < 8; r += 2) {
            float x0 = __shfl_xor(p1[r], 16, 32);
            float x1 = __shfl_xor(p1[r + 1], 16, 32);
            float y0 = __shfl_xor(p0[r], 16, 32);
            float y1 = __shfl_xor(p0[r + 1], 16, 32);
            float a0f = lhi ? x0 : p0[r];
            float a1f = lhi ? x1 : p0[r + 1];
            float b0f = lhi ? p1[r] : y0;
            float b1f = lhi ? p1[r + 1] : y1;
            pf.u[r >> 1]       = pack2((half_t)a0f, (half_t)a1f);
            pf.u[4 + (r >> 1)] = pack2((half_t)b0f, (half_t)b1f);
        }
        g0 = wmma_f16(pf.v, vf[0], g0);
        g1 = wmma_f16(pf.v, vf[1], g1);
        g2 = wmma_f16(pf.v, vf[2], g2);
        g3 = wmma_f16(pf.v, vf[3], g3);
    };

    loadK(0);
    int j0 = 0;
    for (;;) {
        // V for this iteration (consumed at end -> latency hides in softmax)
        v16h vf[4];
        const half_t* vp = vBase + (size_t)lcol * N_ + j0 + (lhi << 4);
        vf[0] = *(const v16h*)(vp);
        vf[1] = *(const v16h*)(vp + (size_t)16 * N_);
        vf[2] = *(const v16h*)(vp + (size_t)32 * N_);
        vf[3] = *(const v16h*)(vp + (size_t)48 * N_);

        const int jb = j0 + (lhi << 3);
        uint64_t mq0 = *(const uint64_t*)(mPad + jb);
        uint64_t mq1 = *(const uint64_t*)(mPad + jb + 16);
        uint32_t m0lo = (uint32_t)mq0, m0hi = (uint32_t)(mq0 >> 32);
        uint32_t m1lo = (uint32_t)mq1, m1hi = (uint32_t)(mq1 >> 32);

        // all 8 S-WMMAs consume kf, then kf is free for in-place prefetch
        v8f STa0 = {}, STa1 = {}, STb0 = {}, STb1 = {};
        STa0 = wmma_f16(kf[0], qB0a, STa0);
        STa0 = wmma_f16(kf[1], qB1a, STa0);
        STa1 = wmma_f16(kf[2], qB0a, STa1);
        STa1 = wmma_f16(kf[3], qB1a, STa1);
        STb0 = wmma_f16(kf[0], qB0b, STb0);
        STb0 = wmma_f16(kf[1], qB1b, STb0);
        STb1 = wmma_f16(kf[2], qB0b, STb1);
        STb1 = wmma_f16(kf[3], qB1b, STb1);

        const bool more = (j0 + 32 <= i0);
        if (more) loadK(j0 + 32);               // prefetch next K in place

        group(STa0, STa1, bRowA, rowOKA, iA, jb, m0lo, m0hi, m1lo, m1hi,
              mA, lA, vf, oa0, oa1, oa2, oa3);
        group(STb0, STb1, bRowB, rowOKB, iB, jb, m0lo, m0hi, m1lo, m1hi,
              mBv, lB, vf, ob0, ob1, ob2, ob3);

        j0 += 32;
        if (!more) break;
    }

    // ---- epilogue ----
    const float invA = 1.0f / lA;
    const float invB = 1.0f / lB;
#pragma unroll
    for (int r = 0; r < 8; ++r) {
        const float iRa = __shfl(invA, r + (lhi << 3), 32);
        const float iRb = __shfl(invB, r + (lhi << 3), 32);
        const int rowA = i0 + r + (lhi << 3);
        const int rowB = rowA + 16;
        half_t* opA = attnh + (size_t)(b * N_ + rowA) * C_ + h * D_ + lcol;
        half_t* opB = attnh + (size_t)(b * N_ + rowB) * C_ + h * D_ + lcol;
        opA[0]  = (half_t)(oa0[r] * iRa);
        opA[16] = (half_t)(oa1[r] * iRa);
        opA[32] = (half_t)(oa2[r] * iRa);
        opA[48] = (half_t)(oa3[r] * iRa);
        opB[0]  = (half_t)(ob0[r] * iRb);
        opB[16] = (half_t)(ob1[r] * iRb);
        opB[32] = (half_t)(ob2[r] * iRb);
        opB[48] = (half_t)(ob3[r] * iRb);
    }
}

// ---------------------------------------------------------------------------
extern "C" void kernel_launch(void* const* d_in, const int* in_sizes, int n_in,
                              void* d_out, int out_size, void* d_ws, size_t ws_size,
                              hipStream_t stream) {
    const float*   x      = (const float*)d_in[0];
    const uint8_t* mask   = (const uint8_t*)d_in[1];
    const float*   bias   = (const float*)d_in[2];
    const float*   W_qkv  = (const float*)d_in[3];
    const float*   W_out  = (const float*)d_in[4];
    const float*   temp   = (const float*)d_in[5];
    float*         out    = (float*)d_out;

    char* ws = (char*)d_ws;
    size_t off = 0;
    auto carve = [&](size_t bytes) {
        char* p = ws + off;
        off += (bytes + 255) & ~(size_t)255;
        return p;
    };
    const size_t M = (size_t)B_ * N_;
    half_t* qkvH  = (half_t*)carve(M * QKV_ * sizeof(half_t));
    half_t* xh    = (half_t*)carve(M * C_ * sizeof(half_t));
    half_t* wqkvh = (half_t*)carve((size_t)QKV_ * C_ * sizeof(half_t));
    half_t* wouth = (half_t*)carve((size_t)C_ * C_ * sizeof(half_t));
    half_t* qh    = (half_t*)carve((size_t)B_ * H_ * N_ * D_ * sizeof(half_t));
    half_t* kh    = (half_t*)carve((size_t)B_ * H_ * N_ * D_ * sizeof(half_t));
    half_t* vt    = (half_t*)carve((size_t)B_ * H_ * N_ * D_ * sizeof(half_t));
    (void)carve(4096);
    half_t* attnh = xh;

    {
        int nx = (int)(M * C_);
        cvt_f32_f16<<<(nx + 255) / 256, 256, 0, stream>>>(x, xh, nx);
        int nw = QKV_ * C_;
        cvt_f32_f16<<<(nw + 255) / 256, 256, 0, stream>>>(W_qkv, wqkvh, nw);
        int no = C_ * C_;
        cvt_f32_f16<<<(no + 255) / 256, 256, 0, stream>>>(W_out, wouth, no);
    }

    {
        int waves = (int)(M / 32) * (QKV_ / 64);       // 9216
        gemm_wmma_ab<C_, half_t><<<waves / 8, 256, 0, stream>>>(xh, wqkvh, qkvH,
                                                                (int)M, QKV_);
    }
    {
        int waves = B_ * H_ * N_;                      // 98304
        qkv_prep<<<waves / 8, 256, 0, stream>>>(qkvH, qh, kh, vt);
    }
    {
        int waves = B_ * H_ * (N_ / 32);               // 3072
        attn_flash<<<waves / 8, 256, 0, stream>>>(qh, kh, vt, bias, mask, temp, attnh);
    }
    {
        int waves = (int)(M / 32) * (C_ / 64);         // 3072
        gemm_wmma_ab<C_, float><<<waves / 8, 256, 0, stream>>>(attnh, wouth, out,
                                                               (int)M, C_);
    }
}
